// VOSRaftUpsampler_38706245272273
// MI455X (gfx1250) — compile-verified, matrix operand order checked by reading, not run on hardware
//
#include <hip/hip_runtime.h>
#include <hip/hip_bf16.h>

typedef _Float16 half_t;
typedef __attribute__((ext_vector_type(16))) _Float16 v16h;
typedef __attribute__((ext_vector_type(8)))  float    v8f;

#define HW 16384
#define IMG 128
#define BSTRIDE 48   // LDS row stride in halves (96B: 32B-aligned, conflict-reducing)

// ---------------- prep kernels ----------------

__global__ void cvt_f32_to_f16_kernel(const float* __restrict__ src,
                                      half_t* __restrict__ dst, int n) {
  int i = blockIdx.x * blockDim.x + threadIdx.x;
  if (i < n) dst[i] = (half_t)src[i];
}

// OIHW f32 -> [oc][tap*IC + ic] f16
__global__ void pack_w3x3_kernel(const float* __restrict__ w,
                                 half_t* __restrict__ wp, int OC, int IC) {
  int t = blockIdx.x * blockDim.x + threadIdx.x;
  int total = OC * IC * 9;
  if (t >= total) return;
  int oc = t / (IC * 9);
  int r = t - oc * IC * 9;
  int ic = r / 9;
  int tap = r - ic * 9;
  wp[(size_t)oc * IC * 9 + tap * IC + ic] = (half_t)w[t];
}

// gw1 [256][512][3][3] f32; fold cat(feat,feat): eff[oc][ic] = gw1[oc][ic]+gw1[oc][ic+256]
// output [oc][tap*256 + ic] f16
__global__ void pack_gw1_kernel(const float* __restrict__ gw1,
                                half_t* __restrict__ wp) {
  int t = blockIdx.x * blockDim.x + threadIdx.x;
  const int total = 256 * 256 * 9;
  if (t >= total) return;
  int oc = t / (256 * 9);
  int r = t - oc * 256 * 9;
  int ic = r / 9;
  int tap = r - ic * 9;
  float v = gw1[((size_t)oc * 512 + ic) * 9 + tap] +
            gw1[((size_t)oc * 512 + ic + 256) * 9 + tap];
  wp[(size_t)oc * 2304 + tap * 256 + ic] = (half_t)v;
}

// ---------------- implicit-GEMM conv via WMMA ----------------
// Block = 256 threads (8 waves). One block: OCB*16 out-channels x 128 pixels
// (one full image row). Wave w owns pixel columns [w*16, w*16+16) and keeps
// OCB f32 accumulators, reusing one staged B tile for OCB back-to-back WMMAs.
// K = ks*ks*IC tiled by 32 (IC is a multiple of 32 everywhere here, so a
// 32-K tile stays inside one conv tap). LDS B stage is double-buffered with
// one barrier per K-tile; the next tile's global loads overlap current WMMAs.
//
// Fill layout: thread owns pixel p = tid&127 and k-group g = tid>>7 (uniform
// per wave). Its 16 elements (ic = icb+g*16..+15 at fixed pixel) are
// k-contiguous in LDS -> one v16h store. Bounds handling is hoisted out of
// the element loop: clamped base address + one exec-masked vector zero.
template <int OCB>
__global__ __launch_bounds__(256)
void conv_wmma_kernel(const half_t* __restrict__ X,   // [N][IC][128][128] f16
                      const half_t* __restrict__ Wp,  // [OC][K] f16 (tap-major,ic-minor)
                      const float* __restrict__ bias, // [OC]
                      void* __restrict__ outp,
                      int IC, int OC, int K, int ks, int outF16, int doRelu) {
  const int tid  = threadIdx.x;
  const int lane = tid & 31;
  const int wave = tid >> 5;
  const int h      = blockIdx.x;              // one image row per block
  const int ocBase = blockIdx.y * (OCB * 16);
  const int n      = blockIdx.z;

  __shared__ __align__(32) half_t Bs[2][128 * BSTRIDE];

  const int pad = ks >> 1;
  const int ntiles = K >> 5;

  v8f acc[OCB] = {};

  const int fp = tid & 127;                   // this thread's pixel
  const int fg = tid >> 7;                    // k-group 0/1 (uniform per wave)

  auto fill = [&](int kt, int buf) {
    const int kbase = kt << 5;
    const int tap = kbase / IC;               // uniform over the tile
    const int icb = kbase - tap * IC;
    const int dy = tap / ks - pad;
    const int dx = tap % ks - pad;
    const int ys = h + dy;
    const bool yok = (unsigned)ys < (unsigned)IMG;
    const int xs = fp + dx;
    const bool ok = yok && (unsigned)xs < (unsigned)IMG;
    const int xsc = xs < 0 ? 0 : (xs > IMG - 1 ? IMG - 1 : xs);
    const int ysc = yok ? ys : 0;
    // 16 unconditional, per-instruction-uniform-k, x-coalesced u16 loads
    const half_t* s2 =
        X + ((size_t)(n * IC + icb + fg * 16)) * HW + ysc * IMG + xsc;
    v16h v;
    #pragma unroll
    for (int i = 0; i < 16; ++i) v[i] = s2[(size_t)i * HW];
    if (!ok) { v16h z = {}; v = z; }          // halo/out-of-image lanes -> 0
    *(v16h*)(&Bs[buf][fp * BSTRIDE + fg * 16]) = v;
    // speculative prefetch of the likely next K-tile rows (same tap, ic+32)
    __builtin_prefetch(s2 + (size_t)32 * HW, 0, 0);
  };

  const int khalf  = (lane >> 4) << 4;        // K half selected by lane[4]
  const int mrow   = lane & 15;               // A: M row ; B: N column
  const int pixcol = (wave << 4) + mrow;

  fill(0, 0);
  for (int kt = 0; kt < ntiles; ++kt) {
    __syncthreads();
    const int kbase = kt << 5;
    // B: 32x16 f16, per-lane contiguous 32B ds load
    v16h b = *(const v16h*)(&Bs[kt & 1][pixcol * BSTRIDE + khalf]);
    #pragma unroll
    for (int j = 0; j < OCB; ++j) {
      // A: 16x32 f16 weights, per-lane contiguous 32B global load (L2-resident)
      v16h a = *(const v16h*)(Wp + (size_t)(ocBase + j * 16 + mrow) * K +
                              kbase + khalf);
      acc[j] = __builtin_amdgcn_wmma_f32_16x16x32_f16(
          false, a, false, b, (short)0, acc[j], false, false);
    }
    if (kt + 1 < ntiles) fill(kt + 1, (kt + 1) & 1);  // overlaps with WMMAs
  }

  // epilogue: bias (+ReLU), store NCHW-flat [n][oc][pix]
  const int pix = h * IMG + pixcol;
  #pragma unroll
  for (int j = 0; j < OCB; ++j) {
    #pragma unroll
    for (int i = 0; i < 8; ++i) {
      int m = ((lane >> 4) << 3) + i;         // C/D layout: VGPR i -> M=(lane/16)*8+i
      int oc = ocBase + j * 16 + m;
      float v = acc[j][i] + bias[oc];
      if (doRelu) v = fmaxf(v, 0.0f);
      size_t o = ((size_t)(n * OC + oc)) * HW + pix;
      if (outF16) ((half_t*)outp)[o] = (half_t)v;
      else        ((float*)outp)[o]  = v;
    }
  }
}

// ---------------- softmax over the 9 window taps ----------------
// wbuf: [N][144][HW] f32, channel = k*16 + s; softmax over k per (n,s,pix)
__global__ void softmax9_kernel(float* __restrict__ wbuf) {
  int t = blockIdx.x * blockDim.x + threadIdx.x;
  if (t >= 2 * 16 * HW) return;
  int pix = t & (HW - 1);
  int s   = (t >> 14) & 15;
  int n   = t >> 18;
  size_t base = ((size_t)(n * 144 + s)) * HW + pix;
  float v[9], m = -1e30f;
  #pragma unroll
  for (int k = 0; k < 9; ++k) {
    v[k] = wbuf[base + (size_t)k * 16 * HW];
    m = fmaxf(m, v[k]);
  }
  float sum = 0.0f;
  #pragma unroll
  for (int k = 0; k < 9; ++k) { v[k] = __expf(v[k] - m); sum += v[k]; }
  float inv = 1.0f / sum;
  #pragma unroll
  for (int k = 0; k < 9; ++k) wbuf[base + (size_t)k * 16 * HW] = v[k] * inv;
}

// ---------------- convex combination + pixel shuffle ----------------
// out[n][c][4h+a][4w+b] = sum_k wts[n][k*16+(a*4+b)][h][w] * seg[n][c][h+kh-1][w+kw-1]
__global__ __launch_bounds__(256)
void combine_kernel(const float* __restrict__ seg,   // [N][64][HW]
                    const float* __restrict__ wts,   // [N][144][HW] (post-softmax)
                    float* __restrict__ out) {       // [N][64][512][512]
  int blk = blockIdx.x;                    // N*HW/4 = 8192 blocks
  int n  = blk >> 12;
  int pb = (blk & 4095) << 2;              // 4 pixels per block
  __shared__ float Ws[4 * 144];
  for (int i = threadIdx.x; i < 576; i += 256) {
    int p = i / 144, ch = i - p * 144;
    Ws[p * 144 + ch] = wts[((size_t)(n * 144 + ch)) * HW + pb + p];
  }
  __syncthreads();
  int p = threadIdx.x >> 6;                // 0..3
  int c = threadIdx.x & 63;
  int pix = pb + p;
  int h = pix >> 7, w = pix & 127;
  const float* sb = seg + ((size_t)(n * 64 + c)) * HW;
  float tap[9];
  #pragma unroll
  for (int k = 0; k < 9; ++k) {
    int yy = h + k / 3 - 1, xx = w + k % 3 - 1;
    tap[k] = ((unsigned)yy < (unsigned)IMG && (unsigned)xx < (unsigned)IMG)
                 ? sb[yy * IMG + xx] : 0.0f;
  }
  float* ob = out + ((size_t)(n * 64 + c)) * 512 * 512;
  const float* wrow = Ws + p * 144;
  #pragma unroll
  for (int a = 0; a < 4; ++a) {            // b128 stores: 4 contiguous floats
    float4 vv;
    float* vp = (float*)&vv;
    #pragma unroll
    for (int b = 0; b < 4; ++b) {
      float acc = 0.0f;
      #pragma unroll
      for (int k = 0; k < 9; ++k) acc += wrow[k * 16 + a * 4 + b] * tap[k];
      vp[b] = acc;
    }
    *(float4*)(ob + (size_t)(4 * h + a) * 512 + 4 * w) = vv;
  }
}

// ---------------- launch ----------------

extern "C" void kernel_launch(void* const* d_in, const int* in_sizes, int n_in,
                              void* d_out, int out_size, void* d_ws, size_t ws_size,
                              hipStream_t stream) {
  (void)in_sizes; (void)n_in; (void)out_size; (void)ws_size;
  const float* x    = (const float*)d_in[0];   // [2][64][128][128]
  const float* feat = (const float*)d_in[1];   // [2][256][128][128]
  const float* w1   = (const float*)d_in[2];   // [32][64][3][3]
  const float* b1   = (const float*)d_in[3];
  const float* w2   = (const float*)d_in[4];   // [64][32][3][3]
  const float* b2   = (const float*)d_in[5];
  const float* gw1  = (const float*)d_in[6];   // [256][512][3][3]
  const float* gb1  = (const float*)d_in[7];
  const float* gw2  = (const float*)d_in[8];   // [144][256][1][1]
  const float* gb2  = (const float*)d_in[9];
  float* out = (float*)d_out;

  // workspace carve-out (all offsets 256B aligned)
  char* ws = (char*)d_ws;
  size_t off = 0;
  half_t* x_f16   = (half_t*)(ws + off); off += (size_t)2 * 64  * HW * 2;
  half_t* feat_f16= (half_t*)(ws + off); off += (size_t)2 * 256 * HW * 2;
  half_t* w1p     = (half_t*)(ws + off); off += (size_t)32  * 576  * 2;
  half_t* w2p     = (half_t*)(ws + off); off += (size_t)64  * 288  * 2;
  half_t* gw1p    = (half_t*)(ws + off); off += (size_t)256 * 2304 * 2;
  half_t* gw2p    = (half_t*)(ws + off); off += (size_t)144 * 256  * 2;
  half_t* seg1    = (half_t*)(ws + off); off += (size_t)2 * 32  * HW * 2;  // relu(conv1)
  float*  segf    = (float*) (ws + off); off += (size_t)2 * 64  * HW * 4;  // conv2 out
  half_t* g1f16   = (half_t*)(ws + off); off += (size_t)2 * 256 * HW * 2;  // relu(gconv1)
  float*  wbuf    = (float*) (ws + off); off += (size_t)2 * 144 * HW * 4;  // gconv2/softmax

  // 1) precision conversion + weight packing (folds cat(feat,feat) into gw1)
  cvt_f32_to_f16_kernel<<<(2*64*HW + 255)/256, 256, 0, stream>>>(x, x_f16, 2*64*HW);
  cvt_f32_to_f16_kernel<<<(2*256*HW + 255)/256, 256, 0, stream>>>(feat, feat_f16, 2*256*HW);
  cvt_f32_to_f16_kernel<<<(144*256 + 255)/256, 256, 0, stream>>>(gw2, gw2p, 144*256);
  pack_w3x3_kernel<<<(32*64*9 + 255)/256, 256, 0, stream>>>(w1, w1p, 32, 64);
  pack_w3x3_kernel<<<(64*32*9 + 255)/256, 256, 0, stream>>>(w2, w2p, 64, 32);
  pack_gw1_kernel<<<(256*256*9 + 255)/256, 256, 0, stream>>>(gw1, gw1p);

  // 2) WMMA convs: grid = (128 rows, OC/(16*OCB), N), block 256 = 8 waves
  conv_wmma_kernel<2><<<dim3(128, 1, 2), 256, 0, stream>>>(
      x_f16,  w1p, b1, seg1, 64,  32,  576,  3, /*f16*/1, /*relu*/1);
  conv_wmma_kernel<4><<<dim3(128, 1, 2), 256, 0, stream>>>(
      seg1,   w2p, b2, segf, 32,  64,  288,  3, /*f16*/0, /*relu*/0);
  conv_wmma_kernel<4><<<dim3(128, 4, 2), 256, 0, stream>>>(
      feat_f16, gw1p, gb1, g1f16, 256, 256, 2304, 3, /*f16*/1, /*relu*/1);
  conv_wmma_kernel<3><<<dim3(128, 3, 2), 256, 0, stream>>>(
      g1f16,  gw2p, gb2, wbuf, 256, 144, 256,  1, /*f16*/0, /*relu*/0);

  // 3) softmax over 9 taps, in place
  softmax9_kernel<<<(2*16*HW + 255)/256, 256, 0, stream>>>(wbuf);

  // 4) weighted 3x3 unfold + 4x pixel shuffle (float4 stores)
  combine_kernel<<<2 * HW / 4, 256, 0, stream>>>(segf, wbuf, out);
}
